// DoubleBlock_20641612825272
// MI455X (gfx1250) — matrix operand order repaired and checked by reading, not compiled, not run on hardware
//
#include <hip/hip_runtime.h>
#include <hip/hip_bf16.h>
#include <cstddef>

// ---------------------------------------------------------------------------
// SwinV2 double block for MI455X (gfx1250): f16 WMMA GEMMs + f32 epilogues,
// with async global->LDS (ASYNCcnt) double-buffered GEMM staging.
// ---------------------------------------------------------------------------

typedef _Float16 half_t;
typedef __attribute__((ext_vector_type(16))) _Float16 v16h;
typedef __attribute__((ext_vector_type(8)))  _Float16 v8h;
typedef __attribute__((ext_vector_type(8)))  float    v8f;
typedef __attribute__((ext_vector_type(4)))  int      v4i;

// Problem constants
static constexpr int Hs   = 56;
static constexpr int Ws   = 56;
static constexpr int WS   = 7;
static constexpr int SHIFT= 3;
static constexpr int NH   = 12;
static constexpr int C    = 192;
static constexpr int DIM  = 2 * C;        // 384
static constexpr int HD   = DIM / NH;     // 32
static constexpr int Ntok = WS * WS;      // 49
static constexpr int NP   = 64;           // padded tokens per window
static constexpr int B    = 8;
static constexpr int NWIN = B * (Hs / WS) * (Ws / WS);   // 512
static constexpr int MW   = NWIN * NP;                   // 32768 padded window rows
static constexpr int MT   = B * Hs * Ws;                 // 25088 tokens
static constexpr int HID  = 4 * DIM;                     // 1536

// ---------------------------------------------------------------------------
// gfx1250 async global->LDS copy (GLOBAL_LOAD_ASYNC_TO_LDS_B128, ASYNCcnt).
// Builtin prototype (from hipcc diagnostic): (v4i as1*, v4i as3*, Ii, Ii).
// Guarded: falls back to load+ds_store if the builtins are absent.
// ---------------------------------------------------------------------------
#if defined(__HIP_DEVICE_COMPILE__) &&                                         \
    __has_builtin(__builtin_amdgcn_global_load_async_to_lds_b128) &&           \
    __has_builtin(__builtin_amdgcn_s_wait_asynccnt)
#define HAVE_ASYNC_LDS 1
#else
#define HAVE_ASYNC_LDS 0
#endif

__device__ inline void async_copy16(const half_t* __restrict__ g, half_t* __restrict__ l) {
#if HAVE_ASYNC_LDS
  __builtin_amdgcn_global_load_async_to_lds_b128(
      (__attribute__((address_space(1))) v4i*)g,
      (__attribute__((address_space(3))) v4i*)l, 0, 0);
#else
  *reinterpret_cast<v8h*>(l) = *reinterpret_cast<const v8h*>(g);
#endif
}

template <int N>
__device__ inline void async_wait() {
#if HAVE_ASYNC_LDS
  __builtin_amdgcn_s_wait_asynccnt(N);
#endif
}

// ---------------------------------------------------------------------------
// WMMA fragment loaders (contiguous k order, same in A and B, wave32).
// A (16x32 f16): lane m=lane&15; contiguous k {kb..kb+15},
//                kb = (lane>=16 ? 16 : 0). Two 16B loads.
// B (32x16 f16): lane n=lane&15; contiguous k {kb2..kb2+15}, kb2 = (lane>=16?16:0).
//                One 32B load from (N,K)-row-major (i.e. B^T / torch weight).
// C/D (16x16 f32): vgpr r, lane: m = r + 8*(lane>>4), n = lane&15.
// ---------------------------------------------------------------------------
__device__ inline v16h load_frag_a(const half_t* __restrict__ base, int ld, int lane) {
  const int m  = lane & 15;
  const int kb = (lane >> 4) << 4;
  const v8h lo = *reinterpret_cast<const v8h*>(base + (size_t)m * ld + kb);
  const v8h hi = *reinterpret_cast<const v8h*>(base + (size_t)m * ld + kb + 8);
  v16h r;
#pragma unroll
  for (int i = 0; i < 8; ++i) { r[i] = lo[i]; r[i + 8] = hi[i]; }
  return r;
}

__device__ inline v16h load_frag_b(const half_t* __restrict__ base, int ld, int lane) {
  const int n   = lane & 15;
  const int kb2 = (lane >> 4) << 4;
  return *reinterpret_cast<const v16h*>(base + (size_t)n * ld + kb2);
}

__device__ inline float gelu_exact(float x) {
  return 0.5f * x * (1.0f + erff(x * 0.70710678118654752f));
}

// ---------------------------------------------------------------------------
// Generic WMMA GEMM: Cout[M,N] = A[M,K] (f16) x W[N,K]^T (f16) + bias
// ACT: 0 = none, 1 = exact GELU.  OUT: 0 = f16 buffer, 1 = f32 + residual.
// Block = 128 threads (4 waves); tile = 64x64; wave w owns rows [w*16, w*16+16).
// K-step 32; A/B tiles double-buffered in LDS via async global->LDS DMA.
// Each thread issues 4 b128 async ops per tile, so asynccnt<=4 waits for the
// current tile while the next tile's DMA is in flight.
// M, N multiples of 64; K multiple of 32.
// ---------------------------------------------------------------------------
template <int ACT, int OUT>
__global__ void gemm_wmma_kernel(const half_t* __restrict__ A,
                                 const half_t* __restrict__ W,
                                 const float*  __restrict__ bias,
                                 half_t*       __restrict__ outH,
                                 float*        __restrict__ outF,
                                 const float*  __restrict__ resid,
                                 int M, int N, int K) {
  const int tid  = threadIdx.x;
  const int lane = tid & 31;
  const int wave = tid >> 5;
  const int row0 = blockIdx.y * 64;
  const int col0 = blockIdx.x * 64;

  __shared__ alignas(32) half_t sA[2][64 * 32];
  __shared__ alignas(32) half_t sB[2][64 * 32];

  // Two 16-byte chunks per matrix per thread: chunk c -> row c/4, half-off (c&3)*8.
  const int c0 = tid * 2, c1 = tid * 2 + 1;
  const int r0 = c0 >> 2, o0 = (c0 & 3) * 8;
  const int r1 = c1 >> 2, o1 = (c1 & 3) * 8;

  auto issue = [&](int it) {
    const int buf = it & 1;
    const int k0  = it * 32;
    async_copy16(A + (size_t)(row0 + r0) * K + k0 + o0, &sA[buf][r0 * 32 + o0]);
    async_copy16(A + (size_t)(row0 + r1) * K + k0 + o1, &sA[buf][r1 * 32 + o1]);
    async_copy16(W + (size_t)(col0 + r0) * K + k0 + o0, &sB[buf][r0 * 32 + o0]);
    async_copy16(W + (size_t)(col0 + r1) * K + k0 + o1, &sB[buf][r1 * 32 + o1]);
  };

  v8f acc[4] = {};
  const int nIter = K / 32;

  issue(0);
  for (int it = 0; it < nIter; ++it) {
    if (it + 1 < nIter) { issue(it + 1); async_wait<4>(); }
    else                { async_wait<0>(); }
    __syncthreads();

    const int buf = it & 1;
    const v16h afrag = load_frag_a(&sA[buf][wave * 16 * 32], 32, lane);
#pragma unroll
    for (int t = 0; t < 4; ++t) {
      const v16h bfrag = load_frag_b(&sB[buf][t * 16 * 32], 32, lane);
      acc[t] = __builtin_amdgcn_wmma_f32_16x16x32_f16(
          false, afrag, false, bfrag, (short)0, acc[t], false, false);
    }
    __syncthreads();   // protects LDS buffer reuse by issue(it+2)
  }

  const int mb = (lane >> 4) * 8;
#pragma unroll
  for (int t = 0; t < 4; ++t) {
    const int n  = col0 + t * 16 + (lane & 15);
    const float bv = bias ? bias[n] : 0.0f;
#pragma unroll
    for (int r = 0; r < 8; ++r) {
      const int m = row0 + wave * 16 + mb + r;
      float v = acc[t][r] + bv;
      if (ACT == 1) v = gelu_exact(v);
      const size_t idx = (size_t)m * N + n;
      if (OUT == 0) outH[idx] = (half_t)v;
      else          outF[idx] = resid[idx] + v;
    }
  }
}

// ---------------------------------------------------------------------------
// LN1 + roll(-3,-3) + window partition. One block per padded window row.
// Pad rows (i >= 49) are zeroed so GEMM sees clean data.
// ---------------------------------------------------------------------------
__global__ void ln1_window_kernel(const float* __restrict__ x0,
                                  const float* __restrict__ x1,
                                  const float* __restrict__ g,
                                  const float* __restrict__ bt,
                                  half_t* __restrict__ xw) {
  const int dest = blockIdx.x;             // 0 .. MW-1
  const int tid  = threadIdx.x;
  const int w    = dest >> 6;
  const int i    = dest & 63;
  half_t* out = xw + (size_t)dest * DIM;
  if (i >= Ntok) {
    for (int c = tid; c < DIM; c += 128) out[c] = (half_t)0.0f;
    return;
  }
  const int b  = w >> 6;
  const int wy = (w >> 3) & 7;
  const int wx = w & 7;
  const int hy = (wy * WS + i / WS + SHIFT) % Hs;   // roll(-3): window reads x[r+3]
  const int hx = (wx * WS + i % WS + SHIFT) % Ws;
  const size_t tok = (size_t)b * (Hs * Ws) + (size_t)hy * Ws + hx;
  const float* p0 = x0 + tok * C;
  const float* p1 = x1 + tok * C;

  __shared__ float r1[128], r2[128];
  float vals[3];
  float s = 0.f, s2 = 0.f;
  for (int c = tid, j = 0; c < DIM; c += 128, ++j) {
    const float v = (c < C) ? p0[c] : p1[c - C];
    vals[j] = v; s += v; s2 += v * v;
  }
  r1[tid] = s; r2[tid] = s2;
  __syncthreads();
  for (int o = 64; o > 0; o >>= 1) {
    if (tid < o) { r1[tid] += r1[tid + o]; r2[tid] += r2[tid + o]; }
    __syncthreads();
  }
  const float mu   = r1[0] * (1.0f / DIM);
  const float var  = r2[0] * (1.0f / DIM) - mu * mu;
  const float rstd = rsqrtf(var + 1e-5f);
  for (int c = tid, j = 0; c < DIM; c += 128, ++j)
    out[c] = (half_t)((vals[j] - mu) * rstd * g[c] + bt[c]);
}

// ---------------------------------------------------------------------------
// Attention for one (window, head): RoPE + L2-normalize q/k, QK^T via WMMA,
// logit-scale + shift mask + softmax (f32), PV via WMMA.
// Block = 128 threads (4 waves).
// ---------------------------------------------------------------------------
__global__ void attn_kernel(const half_t* __restrict__ qkv,
                            const float*  __restrict__ logit_scale,
                            half_t*       __restrict__ outh) {
  const int w    = blockIdx.x;
  const int h    = blockIdx.y;
  const int tid  = threadIdx.x;
  const int lane = tid & 31;
  const int wave = tid >> 5;

  __shared__ alignas(32) half_t sq[NP * HD];    // qn, row-major (64x32)
  __shared__ alignas(32) half_t sk[NP * HD];    // kn, row-major (64x32)
  __shared__ alignas(32) half_t svT[HD * NP];   // v transposed (32x64)
  __shared__ alignas(32) half_t sp[NP * NP];    // softmax probs (64x64)
  __shared__ float  sattn[NP * NP];             // f32 scores
  __shared__ int    sreg[NP];
  __shared__ float  sscale;

  const half_t* base = qkv + (size_t)w * NP * (3 * DIM);

  // V -> LDS transposed (so PV's B-fragment is a contiguous (N,K) load).
  for (int e = tid; e < NP * HD; e += 128) {
    const int i = e >> 5, d = e & 31;
    svT[d * NP + i] = base[(size_t)i * (3 * DIM) + 2 * DIM + h * HD + d];
  }

  if (tid < NP) {
    const int i = tid;
    if (i < Ntok) {
      float qv[HD], kv[HD], qr[HD], kr[HD];
#pragma unroll
      for (int d = 0; d < HD; ++d) {
        qv[d] = (float)base[(size_t)i * (3 * DIM) + h * HD + d];
        kv[d] = (float)base[(size_t)i * (3 * DIM) + DIM + h * HD + d];
      }
      // RoPE: inv_d = 10000^(-d/16) = exp(-d * ln(10000)/16)
#pragma unroll
      for (int d = 0; d < HD / 2; ++d) {
        const float inv = __expf(-(float)d * (9.210340372f / 16.0f));
        float sn, cs;
        __sincosf((float)i * inv, &sn, &cs);
        qr[d]          = qv[d] * cs - qv[d + 16] * sn;
        qr[d + 16]     = qv[d] * sn + qv[d + 16] * cs;
        kr[d]          = kv[d] * cs - kv[d + 16] * sn;
        kr[d + 16]     = kv[d] * sn + kv[d + 16] * cs;
      }
      float qs = 0.f, ks = 0.f;
#pragma unroll
      for (int d = 0; d < HD; ++d) { qs += qr[d] * qr[d]; ks += kr[d] * kr[d]; }
      const float qsc = 1.0f / fmaxf(sqrtf(qs), 1e-12f);
      const float ksc = 1.0f / fmaxf(sqrtf(ks), 1e-12f);
#pragma unroll
      for (int d = 0; d < HD; ++d) {
        sq[i * HD + d] = (half_t)(qr[d] * qsc);
        sk[i * HD + d] = (half_t)(kr[d] * ksc);
      }
      // shift-mask region id for this token
      const int wy = (w >> 3) & 7, wx = w & 7;
      const int hy = wy * WS + i / WS;
      const int hx = wx * WS + i % WS;
      const int ra = (hy < Hs - WS) ? 0 : ((hy < Hs - SHIFT) ? 1 : 2);
      const int rb = (hx < Ws - WS) ? 0 : ((hx < Ws - SHIFT) ? 1 : 2);
      sreg[i] = ra * 3 + rb;
    } else {
#pragma unroll
      for (int d = 0; d < HD; ++d) { sq[i * HD + d] = (half_t)0.f; sk[i * HD + d] = (half_t)0.f; }
      sreg[i] = -1;
    }
  }
  if (tid == 0) sscale = __expf(fminf(logit_scale[h], 4.60517019f)); // ln(100)
  __syncthreads();

  // --- scores: wave owns rows [wave*16, wave*16+16), K = 32 (single WMMA). ---
  {
    v8f acc[4] = {};
    const v16h afrag = load_frag_a(sq + wave * 16 * HD, HD, lane);
#pragma unroll
    for (int t = 0; t < 4; ++t) {
      const v16h bfrag = load_frag_b(sk + t * 16 * HD, HD, lane);
      acc[t] = __builtin_amdgcn_wmma_f32_16x16x32_f16(
          false, afrag, false, bfrag, (short)0, acc[t], false, false);
    }
    const int mb = (lane >> 4) * 8;
#pragma unroll
    for (int t = 0; t < 4; ++t) {
      const int j = t * 16 + (lane & 15);
#pragma unroll
      for (int r = 0; r < 8; ++r) {
        const int i = wave * 16 + mb + r;
        float sc = acc[t][r] * sscale;
        if (j >= Ntok)               sc = -1e30f;        // padded key column
        else if (sreg[i] != sreg[j]) sc += -100.0f;      // cross-region mask
        sattn[i * NP + j] = sc;
      }
    }
  }
  __syncthreads();

  // --- softmax per row (f32) -> sp (f16). ---
  if (tid < NP) {
    const int i = tid;
    float mx = -1e30f;
    for (int j = 0; j < NP; ++j) mx = fmaxf(mx, sattn[i * NP + j]);
    float sum = 0.f;
    for (int j = 0; j < NP; ++j) {
      const float e = __expf(sattn[i * NP + j] - mx);
      sattn[i * NP + j] = e;
      sum += e;
    }
    const float inv = 1.0f / sum;
    for (int j = 0; j < NP; ++j) sp[i * NP + j] = (half_t)(sattn[i * NP + j] * inv);
  }
  __syncthreads();

  // --- out = P(64x64) @ V(64x32): 2 K-steps, 2 N-tiles. ---
  {
    v8f acc[2] = {};
#pragma unroll
    for (int kk = 0; kk < NP; kk += 32) {
      const v16h pfrag = load_frag_a(sp + wave * 16 * NP + kk, NP, lane);
#pragma unroll
      for (int t = 0; t < 2; ++t) {
        const v16h vfrag = load_frag_b(svT + (size_t)(t * 16) * NP + kk, NP, lane);
        acc[t] = __builtin_amdgcn_wmma_f32_16x16x32_f16(
            false, pfrag, false, vfrag, (short)0, acc[t], false, false);
      }
    }
    const int mb = (lane >> 4) * 8;
#pragma unroll
    for (int t = 0; t < 2; ++t) {
      const int d = t * 16 + (lane & 15);
#pragma unroll
      for (int r = 0; r < 8; ++r) {
        const int i = wave * 16 + mb + r;
        outh[(size_t)(w * NP + i) * DIM + h * HD + d] = (half_t)acc[t][r];
      }
    }
  }
}

// ---------------------------------------------------------------------------
// Un-window + roll(+3,+3) + residual (shortcut recomputed from x0/x1) + LN2.
// One block per token. Writes x2 (f32, final residual) and LN2 output (f16).
// ---------------------------------------------------------------------------
__global__ void unwin_res_ln2_kernel(const float* __restrict__ x0,
                                     const float* __restrict__ x1,
                                     const half_t* __restrict__ projh,
                                     const float* __restrict__ g,
                                     const float* __restrict__ bt,
                                     float* __restrict__ x2,
                                     half_t* __restrict__ yh) {
  const int tok = blockIdx.x;              // 0 .. MT-1
  const int tid = threadIdx.x;
  const int b   = tok / (Hs * Ws);
  const int pix = tok % (Hs * Ws);
  const int hy  = pix / Ws, hx = pix % Ws;
  const int ry  = (hy + Hs - SHIFT) % Hs;  // position in rolled image
  const int rx  = (hx + Ws - SHIFT) % Ws;
  const int w   = b * 64 + (ry / WS) * 8 + (rx / WS);
  const int i   = (ry % WS) * WS + (rx % WS);
  const half_t* pr = projh + (size_t)(w * NP + i) * DIM;

  __shared__ float r1[128], r2[128];
  float vals[3];
  float s = 0.f, s2 = 0.f;
  for (int c = tid, j = 0; c < DIM; c += 128, ++j) {
    const float sc = (c < C) ? x0[(size_t)tok * C + c] : x1[(size_t)tok * C + (c - C)];
    const float v  = sc + (float)pr[c];
    vals[j] = v; s += v; s2 += v * v;
    x2[(size_t)tok * DIM + c] = v;
  }
  r1[tid] = s; r2[tid] = s2;
  __syncthreads();
  for (int o = 64; o > 0; o >>= 1) {
    if (tid < o) { r1[tid] += r1[tid + o]; r2[tid] += r2[tid + o]; }
    __syncthreads();
  }
  const float mu   = r1[0] * (1.0f / DIM);
  const float var  = r2[0] * (1.0f / DIM) - mu * mu;
  const float rstd = rsqrtf(var + 1e-5f);
  for (int c = tid, j = 0; c < DIM; c += 128, ++j)
    yh[(size_t)tok * DIM + c] = (half_t)((vals[j] - mu) * rstd * g[c] + bt[c]);
}

// ---------------------------------------------------------------------------
__global__ void cvt_f2h_kernel(const float* __restrict__ s, half_t* __restrict__ d, int n) {
  for (int i = blockIdx.x * blockDim.x + threadIdx.x; i < n; i += gridDim.x * blockDim.x)
    d[i] = (half_t)s[i];
}

// ---------------------------------------------------------------------------
extern "C" void kernel_launch(void* const* d_in, const int* in_sizes, int n_in,
                              void* d_out, int out_size, void* d_ws, size_t ws_size,
                              hipStream_t stream) {
  const float* x0          = (const float*)d_in[0];
  const float* x1          = (const float*)d_in[1];
  const float* norm1_g     = (const float*)d_in[2];
  const float* norm1_b     = (const float*)d_in[3];
  const float* qkv_w       = (const float*)d_in[4];
  const float* qkv_b       = (const float*)d_in[5];
  const float* logit_scale = (const float*)d_in[6];
  const float* proj_w      = (const float*)d_in[7];
  const float* proj_b      = (const float*)d_in[8];
  const float* norm2_g     = (const float*)d_in[9];
  const float* norm2_b     = (const float*)d_in[10];
  const float* fc1_w       = (const float*)d_in[11];
  const float* fc1_b       = (const float*)d_in[12];
  const float* fc2_w       = (const float*)d_in[13];
  const float* fc2_b       = (const float*)d_in[14];

  char* ws = (char*)d_ws;
  size_t off = 0;
  auto take = [&](size_t bytes) -> void* {
    void* p = ws + off;
    off += (bytes + 255) & ~(size_t)255;
    return p;
  };

  half_t* qkv_wh = (half_t*)take((size_t)3 * DIM * DIM * 2);
  half_t* proj_wh= (half_t*)take((size_t)DIM * DIM * 2);
  half_t* fc1_wh = (half_t*)take((size_t)HID * DIM * 2);
  half_t* fc2_wh = (half_t*)take((size_t)DIM * HID * 2);
  half_t* xw_h   = (half_t*)take((size_t)MW * DIM * 2);
  half_t* qkv_h  = (half_t*)take((size_t)MW * 3 * DIM * 2);
  half_t* attn_h = (half_t*)take((size_t)MW * DIM * 2);
  half_t* proj_h = (half_t*)take((size_t)MW * DIM * 2);
  float*  x2     = (float*) take((size_t)MT * DIM * 4);
  half_t* y_h    = (half_t*)take((size_t)MT * DIM * 2);
  half_t* h_h    = (half_t*)take((size_t)MT * HID * 2);
  (void)ws_size; (void)n_in; (void)in_sizes; (void)out_size;

  // 0) weights -> f16
  cvt_f2h_kernel<<<256, 256, 0, stream>>>(qkv_w, qkv_wh, 3 * DIM * DIM);
  cvt_f2h_kernel<<<256, 256, 0, stream>>>(proj_w, proj_wh, DIM * DIM);
  cvt_f2h_kernel<<<256, 256, 0, stream>>>(fc1_w, fc1_wh, HID * DIM);
  cvt_f2h_kernel<<<256, 256, 0, stream>>>(fc2_w, fc2_wh, DIM * HID);

  // 1) LN1 + shift + window partition
  ln1_window_kernel<<<MW, 128, 0, stream>>>(x0, x1, norm1_g, norm1_b, xw_h);

  // 2) QKV GEMM: (32768 x 384) x (384 x 1152)
  gemm_wmma_kernel<0, 0><<<dim3((3 * DIM) / 64, MW / 64), 128, 0, stream>>>(
      xw_h, qkv_wh, qkv_b, qkv_h, nullptr, nullptr, MW, 3 * DIM, DIM);

  // 3) attention (RoPE + cosine attn + mask + softmax + PV)
  attn_kernel<<<dim3(NWIN, NH), 128, 0, stream>>>(qkv_h, logit_scale, attn_h);

  // 4) proj GEMM: (32768 x 384) x (384 x 384)
  gemm_wmma_kernel<0, 0><<<dim3(DIM / 64, MW / 64), 128, 0, stream>>>(
      attn_h, proj_wh, proj_b, proj_h, nullptr, nullptr, MW, DIM, DIM);

  // 5) un-window + unshift + residual + LN2
  unwin_res_ln2_kernel<<<MT, 128, 0, stream>>>(x0, x1, proj_h, norm2_g, norm2_b, x2, y_h);

  // 6) fc1 GEMM + GELU: (25088 x 384) x (384 x 1536)
  gemm_wmma_kernel<1, 0><<<dim3(HID / 64, MT / 64), 128, 0, stream>>>(
      y_h, fc1_wh, fc1_b, h_h, nullptr, nullptr, MT, HID, DIM);

  // 7) fc2 GEMM + residual -> f32 output: (25088 x 1536) x (1536 x 384)
  gemm_wmma_kernel<0, 1><<<dim3(DIM / 64, MT / 64), 128, 0, stream>>>(
      h_h, fc2_wh, fc2_b, nullptr, (float*)d_out, x2, MT, DIM, HID);
}
